// rnnCell_unrectified_51917564674726
// MI455X (gfx1250) — matrix operand order repaired and checked by reading, not compiled, not run on hardware
//
#include <hip/hip_runtime.h>

#define B_DIM 32768
#define H_DIM 512
#define K_DIM 512
#define BHTOT ((size_t)B_DIM * (size_t)H_DIM)
#define HH    (H_DIM * H_DIM)

typedef __attribute__((ext_vector_type(16))) _Float16 v16h;
typedef __attribute__((ext_vector_type(8)))  _Float16 h8;
typedef __attribute__((ext_vector_type(4)))  _Float16 h4;
typedef __attribute__((ext_vector_type(8)))  float    v8f;

__device__ __forceinline__ v8f wmma16(v16h a, v16h b, v8f c) {
  // D = A(16x32 f16) * B(32x16 f16) + C(16x16 f32)
  return __builtin_amdgcn_wmma_f32_16x16x32_f16(false, a, false, b, (short)0, c, false, false);
}

__device__ __forceinline__ float sigm(float x) { return 1.0f / (1.0f + __expf(-x)); }

__device__ __forceinline__ float4 ld4(const float* p) {
  return *reinterpret_cast<const float4*>(p);
}

__device__ __forceinline__ h4 cvt4(float a, float b, float c, float d) {
  h4 r = { (_Float16)a, (_Float16)b, (_Float16)c, (_Float16)d };
  return r;
}

// A-matrix fragment (16x32 f16): lane L, row M = L%16.
// lanes 0-15: K = {0..15};  lanes 16-31: K = {16..31}
__device__ __forceinline__ v16h fragA(const _Float16* row, int koff) {
  h8 lo = *reinterpret_cast<const h8*>(row + koff);
  h8 hi = *reinterpret_cast<const h8*>(row + 8 + koff);
  return __builtin_shufflevector(lo, hi, 0,1,2,3,4,5,6,7,8,9,10,11,12,13,14,15);
}

// ---------------------------------------------------------------------------
// Kernel P: pack all six weight matrices to f16 (Wzx,Wr,Wbx1,Wby1,Wba1,
//           exp(log_Way)) and zero the rowNorm accumulators.
// ---------------------------------------------------------------------------
__global__ void prep_kernel(const float* __restrict__ Wzx,  const float* __restrict__ Wr,
                            const float* __restrict__ Wbx1, const float* __restrict__ Wby1,
                            const float* __restrict__ Wba1, const float* __restrict__ logWay,
                            _Float16* __restrict__ wh,
                            float* __restrict__ rowNorm) {
  int t = blockIdx.x * blockDim.x + threadIdx.x;
  if (t < HH) {
    wh[t         ] = (_Float16)Wzx[t];
    wh[t + 1 * HH] = (_Float16)Wr[t];
    wh[t + 2 * HH] = (_Float16)Wbx1[t];
    wh[t + 3 * HH] = (_Float16)Wby1[t];
    wh[t + 4 * HH] = (_Float16)Wba1[t];
    wh[t + 5 * HH] = (_Float16)__expf(logWay[t]);
  }
  if (t < B_DIM) rowNorm[t] = 0.0f;
}

// ---------------------------------------------------------------------------
// Kernel D: fused 6-product WMMA GEMM + elementwise update for y,a,b1 and
//           per-row sum(z^2) accumulation (for the B0 gate).
// Block = 256 threads (8 waves), tile = 64 rows x 64 cols, KT = 32.
// Weight tiles are staged with global_load_async_to_lds_b128 (ASYNCcnt).
// ---------------------------------------------------------------------------
__global__ void __launch_bounds__(256) fused_kernel(
    const float* __restrict__ xg,  const float* __restrict__ yg,
    const float* __restrict__ ag,  const float* __restrict__ b0g,
    const float* __restrict__ b1g,
    const _Float16* __restrict__ wh,   // 6 packed f16 weight matrices
    const float* __restrict__ sigma,
    const float* __restrict__ pty,  const float* __restrict__ pta,
    const float* __restrict__ ptb1,
    float* __restrict__ rowNorm,
    float* __restrict__ outY, float* __restrict__ outA, float* __restrict__ outB1)
{
  __shared__ __attribute__((aligned(64))) _Float16 sA[4][64][32]; // x, y, a, norm_in
  __shared__ __attribute__((aligned(64))) _Float16 sB[6][64][32]; // Wzx,Wr,Wbx1,Wby1,Wba1,eWay

  const int tid     = threadIdx.x;
  const int colBase = blockIdx.x * 64;
  const int rowBase = blockIdx.y * 64;
  const int lane    = tid & 31;
  const int wave    = tid >> 5;
  const int lh      = lane & 15;
  const int hiHalf  = (lane >> 4) & 1;
  const int mi      = wave & 3;             // row-subtile 0..3
  const int nbase   = (wave >> 2) * 2;      // col-subtile base 0 or 2
  const int koffA   = hiHalf ? 16 : 0;
  const int koffB   = hiHalf ? 16 : 0;
  const int mrow    = mi * 16 + lh;

  // LDS byte offset of sB (flat pointer low 32 bits == LDS offset)
  const unsigned ldsB0 = (unsigned)(size_t)&sB[0][0][0];
  // this thread's 16B chunk of each 64x32-half weight tile
  const int wr = tid >> 2;                  // tile row 0..63
  const int wq = tid & 3;                   // 16-byte chunk 0..3
  const unsigned lOff0 = ldsB0 + (unsigned)(wr * 64 + wq * 16);

  const v8f vzero = {0.f,0.f,0.f,0.f,0.f,0.f,0.f,0.f};
  v8f accZ[2], accY[2], accB[2], accW[2];
#pragma unroll
  for (int t = 0; t < 2; ++t) { accZ[t]=vzero; accY[t]=vzero; accB[t]=vzero; accW[t]=vzero; }

  for (int kb = 0; kb < K_DIM; kb += 32) {
    // ---- stage weight tiles: async f16 global -> LDS (no VALU) ----
    {
      const unsigned gOff0 =
          (unsigned)((((colBase + wr) * K_DIM + kb) * 2) + wq * 16); // bytes
#pragma unroll
      for (int t = 0; t < 6; ++t) {
        asm volatile("global_load_async_to_lds_b128 %0, %1, %2 offset:0"
                     :: "v"(lOff0 + (unsigned)(t * 4096)),
                        "v"(gOff0 + (unsigned)(t * HH * 2)),
                        "s"(wh)
                     : "memory");
      }
    }
    // ---- stage activation tiles: fp32 global -> f16 LDS (+ norm_in) ----
#pragma unroll
    for (int it = 0; it < 2; ++it) {
      int e4 = tid + it * 256;          // 0..511 (64 rows * 8 float4)
      int r  = e4 >> 3;
      int c4 = (e4 & 7) << 2;
      size_t gA = (size_t)(rowBase + r) * K_DIM + kb + c4;
      float4 xv = ld4(xg + gA);
      float4 yv = ld4(yg + gA);
      float4 av = ld4(ag + gA);
      *reinterpret_cast<h4*>(&sA[0][r][c4]) = cvt4(xv.x, xv.y, xv.z, xv.w);
      *reinterpret_cast<h4*>(&sA[1][r][c4]) = cvt4(yv.x, yv.y, yv.z, yv.w);
      *reinterpret_cast<h4*>(&sA[2][r][c4]) = cvt4(av.x, av.y, av.z, av.w);
      *reinterpret_cast<h4*>(&sA[3][r][c4]) = cvt4(yv.x*yv.x*fmaxf(av.x,0.f),
                                                   yv.y*yv.y*fmaxf(av.y,0.f),
                                                   yv.z*yv.z*fmaxf(av.z,0.f),
                                                   yv.w*yv.w*fmaxf(av.w,0.f));
      if (kb + 32 < K_DIM) {               // global_prefetch_b8 for next k-tile
        __builtin_prefetch(xg + gA + 32, 0, 0);
        __builtin_prefetch(yg + gA + 32, 0, 0);
        __builtin_prefetch(ag + gA + 32, 0, 0);
      }
    }
    asm volatile("s_wait_asynccnt 0x0" ::: "memory");
    __syncthreads();

    // ---- WMMA compute ----
    v16h fx = fragA(&sA[0][mrow][0], koffA);
    v16h fy = fragA(&sA[1][mrow][0], koffA);
    v16h fa = fragA(&sA[2][mrow][0], koffA);
    v16h fn = fragA(&sA[3][mrow][0], koffA);
#pragma unroll
    for (int t = 0; t < 2; ++t) {
      int n = (nbase + t) * 16 + lh;
      v16h fWz = *reinterpret_cast<const v16h*>(&sB[0][n][koffB]);
      v16h fWr = *reinterpret_cast<const v16h*>(&sB[1][n][koffB]);
      v16h fW1 = *reinterpret_cast<const v16h*>(&sB[2][n][koffB]);
      v16h fW2 = *reinterpret_cast<const v16h*>(&sB[3][n][koffB]);
      v16h fW3 = *reinterpret_cast<const v16h*>(&sB[4][n][koffB]);
      v16h fEW = *reinterpret_cast<const v16h*>(&sB[5][n][koffB]);
      accZ[t] = wmma16(fx, fWz, accZ[t]);   // z        = x  @ Wzx^T
      accY[t] = wmma16(fy, fWr, accY[t]);   // y_hat    = y  @ Wr^T
      accB[t] = wmma16(fx, fW1, accB[t]);   // B1 preact: x @ Wbx1^T
      accB[t] = wmma16(fy, fW2, accB[t]);   //          + y @ Wby1^T
      accB[t] = wmma16(fa, fW3, accB[t]);   //          + a @ Wba1^T
      accW[t] = wmma16(fn, fEW, accW[t]);   // norm_in @ exp(log_Way)^T
    }
    __syncthreads();
  }

  // ---- epilogue: elementwise update + row-norm partials ----
  const int rbase = rowBase + mi * 16 + hiHalf * 8;
  float sq[8] = {0.f,0.f,0.f,0.f,0.f,0.f,0.f,0.f};
#pragma unroll
  for (int t = 0; t < 2; ++t) {
    const int j = colBase + (nbase + t) * 16 + lh;
    const float dty  = 0.01f * sigm(pty[j]);
    const float dta  = 0.01f * sigm(pta[j]);
    const float dtb1 = 0.1f  * sigm(ptb1[j]);
    const float sgv  = sigma[j];
    const float sg2  = sgv * sgv;
#pragma unroll
    for (int r = 0; r < 8; ++r) {
      const int i = rbase + r;
      const size_t idx = (size_t)i * H_DIM + j;
      const float zv  = accZ[t][r];
      const float yv  = yg[idx];
      const float av  = ag[idx];
      const float b0v = b0g[idx];
      const float b1v = b1g[idx];
      sq[r] += zv * zv;
      const float B1  = sigm(accB[t][r]);
      outB1[idx] = b1v + dtb1 * (B1 - b1v);
      const float act = sqrtf(fmaxf(av, 0.f));
      outY[idx]  = yv + dty * (b1v * zv + (1.f - act) * accY[t][r] - yv);
      outA[idx]  = av + dta * (sg2 * b0v * b0v + accW[t][r] - av);
    }
  }
  // reduce sum(z^2) across the 16 lanes that share a row, then one atomic/row
#pragma unroll
  for (int r = 0; r < 8; ++r) {
    float s = sq[r];
    s += __shfl_xor(s, 1, 32);
    s += __shfl_xor(s, 2, 32);
    s += __shfl_xor(s, 4, 32);
    s += __shfl_xor(s, 8, 32);
    if (lh == 0) atomicAdd(&rowNorm[rbase + r], s);
  }
}

// ---------------------------------------------------------------------------
// Kernel E: b_new0 = b0 + dt_b0 * (sigmoid(1/(||z||+eps)) - b0)
// ---------------------------------------------------------------------------
__global__ void b0_kernel(const float* __restrict__ b0,
                          const float* __restrict__ rowNorm,
                          const float* __restrict__ ptb0,
                          float* __restrict__ outB0) {
  int t = blockIdx.x * blockDim.x + threadIdx.x;
  int i = t >> 9;            // H_DIM == 512
  int j = t & (H_DIM - 1);
  float B0   = sigm(1.0f / (sqrtf(rowNorm[i]) + 1e-8f));
  float dtb0 = 0.1f * sigm(ptb0[j]);
  float v    = b0[t];
  outB0[t]   = v + dtb0 * (B0 - v);
}

// ---------------------------------------------------------------------------
extern "C" void kernel_launch(void* const* d_in, const int* in_sizes, int n_in,
                              void* d_out, int out_size, void* d_ws, size_t ws_size,
                              hipStream_t stream) {
  (void)in_sizes; (void)n_in; (void)out_size; (void)ws_size;

  const float* x      = (const float*)d_in[0];
  const float* y      = (const float*)d_in[1];
  const float* a      = (const float*)d_in[2];
  const float* b0     = (const float*)d_in[3];
  const float* b1     = (const float*)d_in[4];
  const float* Wzx    = (const float*)d_in[5];
  const float* Wr     = (const float*)d_in[6];
  const float* Wbx1   = (const float*)d_in[7];
  const float* Wby1   = (const float*)d_in[8];
  const float* Wba1   = (const float*)d_in[9];
  const float* logWay = (const float*)d_in[10];
  const float* sigma  = (const float*)d_in[11];
  const float* pty    = (const float*)d_in[12];
  const float* pta    = (const float*)d_in[13];
  const float* ptb0   = (const float*)d_in[14];
  const float* ptb1   = (const float*)d_in[15];

  _Float16* wh      = (_Float16*)d_ws;                              // 6*H*H f16 = 3 MB
  float*    rowNorm = (float*)((char*)d_ws + (size_t)6 * HH * 2);   // B floats = 128 KB

  float* outY  = (float*)d_out;
  float* outA  = outY  + BHTOT;
  float* outB0 = outA  + BHTOT;
  float* outB1 = outB0 + BHTOT;

  prep_kernel<<<HH / 256, 256, 0, stream>>>(Wzx, Wr, Wbx1, Wby1, Wba1, logWay,
                                            wh, rowNorm);

  dim3 grid(H_DIM / 64, B_DIM / 64);   // (8, 512)
  fused_kernel<<<grid, 256, 0, stream>>>(x, y, a, b0, b1, wh,
                                         sigma, pty, pta, ptb1,
                                         rowNorm, outY, outA, outB1);

  b0_kernel<<<(unsigned)(BHTOT / 256), 256, 0, stream>>>(b0, rowNorm, ptb0, outB0);
}
